// DynamicSparseAttention_84567906058559
// MI455X (gfx1250) — matrix operand order repaired and checked by reading, not compiled, not run on hardware
//
#include <hip/hip_runtime.h>

#define B_ 2
#define L_ 2048
#define S_ 2048
#define H_ 16
#define E_ 64
#define D_ 64
#define ST_ 64                    // S-tile (columns per iteration)
#define L2E 1.4426950408889634f

typedef __attribute__((ext_vector_type(16))) __bf16 v16bf;
typedef __attribute__((ext_vector_type(2)))  __bf16 v2bf;
typedef __attribute__((ext_vector_type(8)))  float  v8f;
typedef __attribute__((ext_vector_type(2)))  float  v2f;

union Frag  { v16bf v; unsigned u[8]; };
union FragF { v8f   v; float    f[8]; };

// native f32 -> packed bf16 pair (v_cvt_pk_bf16_f32 when available)
__device__ __forceinline__ unsigned pk2(float a, float b) {
  v2f t; t.x = a; t.y = b;
  return __builtin_bit_cast(unsigned, __builtin_convertvector(t, v2bf));
}

__global__ __launch_bounds__(128)
void dsattn_kernel(const float* __restrict__ Q,
                   const float* __restrict__ K,
                   const float* __restrict__ V,
                   float* __restrict__ O) {
  // K tile: [s=64][e-pair=32], pitch 33 -> conflict-free frag reads & writes
  __shared__ unsigned lK[64 * 33];
  // V tile transposed: [d=64][s-pair=32], pitch 33
  __shared__ unsigned lV[64 * 33];

  const int tid  = threadIdx.x;
  const int lane = tid & 31;
  const int wave = tid >> 5;
  const int m    = lane & 15;   // fragment row/col index; also this lane's q-row
  const int g    = lane >> 4;   // lane-half group

  const int b = blockIdx.z;
  const int h = blockIdx.y;
  const int rowBase = blockIdx.x * 64 + wave * 16;

  const float* Qrow = Q + ((size_t)(b * L_ + rowBase + m) * H_ + h) * E_;
  const float* Kh   = K + ((size_t)(b * S_) * H_ + h) * E_;
  const float* Vh   = V + ((size_t)(b * S_) * H_ + h) * D_;

  // ---- Q as B-fragments of Q^T (B[k=e][n=q-row]), loop-resident
  Frag qb[2];
#pragma unroll
  for (int kf = 0; kf < 2; ++kf)
#pragma unroll
    for (int i = 0; i < 8; ++i) {
      int e0 = 32 * kf + 16 * g + 2 * i;
      qb[kf].u[i] = pk2(Qrow[e0], Qrow[e0 + 1]);
    }

  FragF acc1[4], acc5[4];   // O accumulators (standard C layout), 4 D-tiles
#pragma unroll
  for (int f = 0; f < 4; ++f)
#pragma unroll
    for (int i = 0; i < 8; ++i) { acc1[f].f[i] = 0.f; acc5[f].f[i] = 0.f; }

  // per-lane row statistics (row = lane&15, duplicated across lane halves)
  float mrun  = -INFINITY;
  float zrun1 = 0.f, zrun5 = 0.f;

  for (int s0 = 0; s0 < S_; s0 += ST_) {
    __syncthreads();   // previous tile's LDS reads finished

    const float* Kt = Kh + (size_t)s0 * (H_ * E_);
    const float* Vt = Vh + (size_t)s0 * (H_ * D_);
    if (s0 + ST_ < S_) {  // prefetch next tile into L2 (global_prefetch_b8)
      __builtin_prefetch(Kt + ST_ * (H_ * E_) + tid * 16, 0, 1);
      __builtin_prefetch(Vt + ST_ * (H_ * D_) + tid * 16, 0, 1);
    }

    // cooperative K tile -> LDS as bf16 pairs (coalesced float2 reads)
#pragma unroll 4
    for (int j = 0; j < 16; ++j) {
      int idx = j * 128 + tid;                // 0..2047
      int s = idx >> 5, ep = idx & 31;
      const float* src = Kt + (size_t)s * (H_ * E_) + ep * 2;
      lK[s * 33 + ep] = pk2(src[0], src[1]);
    }
    // cooperative V tile -> LDS transposed: [d][s-pair]
#pragma unroll 4
    for (int j = 0; j < 16; ++j) {
      int idx = j * 128 + tid;
      int d = idx & 63, sp = idx >> 6;        // sp: 0..31
      float a = Vt[(size_t)(2 * sp) * (H_ * D_) + d];
      float c = Vt[(size_t)(2 * sp + 1) * (H_ * D_) + d];
      lV[d * 33 + sp] = pk2(a, c);
    }
    __syncthreads();

    // ---- scores TRANSPOSED: sc[nt] = K_tile(A, 16 s-rows) x Q^T(B)
    //      element (i,lane): s = 16*nt + i + 8*g, q-row = lane&15
    FragF sc[4];
#pragma unroll
    for (int nt = 0; nt < 4; ++nt) {
      v8f c;
#pragma unroll
      for (int i = 0; i < 8; ++i) c[i] = 0.f;
#pragma unroll
      for (int kf = 0; kf < 2; ++kf) {
        Frag ka;
#pragma unroll
        for (int i = 0; i < 8; ++i) {
          int ep = 16 * kf + 8 * g + i;
          ka.u[i] = lK[(16 * nt + m) * 33 + ep];
        }
        c = __builtin_amdgcn_wmma_f32_16x16x32_bf16(false, ka.v, false, qb[kf].v,
                                                    (short)0, c, false, false);
      }
      sc[nt].v = c;
    }

    // ---- per-lane row stats: in-lane max over 32 s-values + one half-swap
    float t = sc[0].f[0];
#pragma unroll
    for (int nt = 0; nt < 4; ++nt)
#pragma unroll
      for (int i = 0; i < 8; ++i) t = fmaxf(t, sc[nt].f[i]);
    t = fmaxf(t, __shfl_xor(t, 16, 32));
    float mn = fmaxf(mrun, t);
    float s1 = __builtin_exp2f((mrun - mn) * L2E);   // e^(m_old-m_new)
    float s2 = s1 * s1;
    float s5 = s2 * s2 * s1;                         // s1^5
    mrun = mn;

    // broadcast per-row scales into accumulator layout (row = i + 8g)
    float s1b[8], s5b[8];
#pragma unroll
    for (int i = 0; i < 8; ++i) {
      float x = __shfl(s1, i + 8 * g, 32);
      float x2 = x * x;
      s1b[i] = x;
      s5b[i] = x2 * x2 * x;
    }
#pragma unroll
    for (int f = 0; f < 4; ++f)
#pragma unroll
      for (int i = 0; i < 8; ++i) {
        acc1[f].f[i] *= s1b[i];
        acc5[f].f[i] *= s5b[i];
      }

    // ---- e / e^5, in-register repack to A-fragments, PV WMMAs
    float zt1 = 0.f, zt5 = 0.f;
#pragma unroll
    for (int kf = 0; kf < 2; ++kf) {
      FragF e1[2], e5[2];
#pragma unroll
      for (int q = 0; q < 2; ++q)
#pragma unroll
        for (int i = 0; i < 8; ++i) {
          float e  = __builtin_exp2f((sc[2 * kf + q].f[i] - mrun) * L2E);
          float ee = e * e;
          float e5v = ee * ee * e;
          e1[q].f[i] = e;
          e5[q].f[i] = e5v;
          zt1 += e;
          zt5 += e5v;
        }
      // Sᵀ C-layout -> P A-fragment: pure in-lane repack (verified both halves)
      Frag pa1, pa5;
#pragma unroll
      for (int i = 0; i < 8; ++i) {
        int q = i >> 2, j = i & 3;
        pa1.u[i] = pk2(e1[q].f[2 * j], e1[q].f[2 * j + 1]);
        pa5.u[i] = pk2(e5[q].f[2 * j], e5[q].f[2 * j + 1]);
      }
#pragma unroll
      for (int f = 0; f < 4; ++f) {
        Frag va;
#pragma unroll
        for (int i = 0; i < 8; ++i)
          va.u[i] = lV[(16 * f + m) * 33 + 16 * kf + ((i < 4) ? 4 * g + i : 8 + 4 * g + (i - 4))];
        acc1[f].v = __builtin_amdgcn_wmma_f32_16x16x32_bf16(false, pa1.v, false, va.v,
                                                            (short)0, acc1[f].v, false, false);
        acc5[f].v = __builtin_amdgcn_wmma_f32_16x16x32_bf16(false, pa5.v, false, va.v,
                                                            (short)0, acc5[f].v, false, false);
      }
    }
    // complete row sums (halves hold disjoint s-ranges) and fold into running Z
    zt1 += __shfl_xor(zt1, 16, 32);
    zt5 += __shfl_xor(zt5, 16, 32);
    zrun1 = zrun1 * s1 + zt1;
    zrun5 = zrun5 * s5 + zt5;
  }

  // ---- finalize: out = (Z^-5*acc5 + Z^-1*acc1) / (Z5*Z^-5 + 1)
#pragma unroll
  for (int i = 0; i < 8; ++i) {
    float Z1 = __shfl(zrun1, i + 8 * g, 32);   // stats for row i+8g
    float Z5 = __shfl(zrun5, i + 8 * g, 32);
    float r1 = 1.0f / Z1;
    float r2 = r1 * r1;
    float r5 = r2 * r2 * r1;
    float dn = 1.0f / (Z5 * r5 + 1.0f);
    size_t row = (size_t)(b * L_) + rowBase + i + 8 * g;
#pragma unroll
    for (int f = 0; f < 4; ++f) {
      float o = (acc5[f].f[i] * r5 + acc1[f].f[i] * r1) * dn;
      O[(row * H_ + h) * D_ + 16 * f + m] = o;
    }
  }
}

extern "C" void kernel_launch(void* const* d_in, const int* in_sizes, int n_in,
                              void* d_out, int out_size, void* d_ws, size_t ws_size,
                              hipStream_t stream) {
  (void)in_sizes; (void)n_in; (void)out_size; (void)d_ws; (void)ws_size;
  const float* Q = (const float*)d_in[0];
  const float* K = (const float*)d_in[1];
  const float* V = (const float*)d_in[2];
  float* O = (float*)d_out;
  dim3 grid(L_ / 64, H_, B_);   // 32 row-tiles x 16 heads x 2 batch
  dim3 block(128);              // 4 waves (wave32)
  dsattn_kernel<<<grid, block, 0, stream>>>(Q, K, V, O);
}
